// sample_net_transpose_61710090109000
// MI455X (gfx1250) — matrix operand order repaired and checked by reading, not compiled, not run on hardware
//
#include <hip/hip_runtime.h>

// gfx1250 WMMA vector types
typedef __attribute__((ext_vector_type(16))) __bf16 v16bf;
typedef __attribute__((ext_vector_type(8)))  float  v8f;
typedef __attribute__((ext_vector_type(4)))  float  v4f;
typedef __attribute__((ext_vector_type(4)))  __bf16 v4bf;
typedef __attribute__((ext_vector_type(4)))  unsigned short us4;

struct Bits32B { uint4 lo; uint4 hi; };   // 32 bytes == one v16bf fragment

__device__ __forceinline__ us4 cvt4_bf16(float a, float b, float c, float d) {
  v4f f = {a, b, c, d};
  v4bf h = __builtin_convertvector(f, v4bf);   // native bf16 cvt (RNE) on gfx1250
  return __builtin_bit_cast(us4, h);
}

#define LDK 72            // padded K stride (elements): 144B rows -> 16B-aligned b128 loads
#define XS_ROWS 128       // n rows
#define WS_ROWS 128       // o rows
#define TILE_BYTES ((XS_ROWS + WS_ROWS) * LDK * 2)   // 36864 B < 64KB

__global__ __launch_bounds__(256)
void conv1x1_d2s_wmma_bf16(const float* __restrict__ x,   // [32,256,32,32]
                           const float* __restrict__ w,   // [1024,256]
                           float* __restrict__ out) {     // [32,1,1024,1024]
  __shared__ __align__(16) unsigned char smemRaw[TILE_BYTES];
  unsigned short (*XsLds)[LDK] =
      reinterpret_cast<unsigned short(*)[LDK]>(smemRaw);                    // [n][k]
  unsigned short (*WsLds)[LDK] =
      reinterpret_cast<unsigned short(*)[LDK]>(smemRaw + XS_ROWS * LDK * 2);// [o][k]
  float* obuf = reinterpret_cast<float*>(smemRaw);   // aliased after compute

  const int tid  = threadIdx.x;
  const int lane = tid & 31;
  const int wave = tid >> 5;
  const int half = lane >> 4;   // 0: lanes 0-15, 1: lanes 16-31
  const int lrow = lane & 15;
  const int wn   = wave & 3;    // 4 waves along n (spatial)
  const int wo   = wave >> 2;   // 2 waves along o (channels)

  const int oBase = blockIdx.x * 128;   // 8 o-tiles
  const int nBase = blockIdx.y * 128;   // 8 n-tiles per batch
  const int b     = blockIdx.z;         // 32 batches

  const size_t xBatch = (size_t)b * 256 * 1024;

  v8f acc[2][4];
  {
    v8f zero;
    #pragma unroll
    for (int e = 0; e < 8; ++e) zero[e] = 0.0f;
    #pragma unroll
    for (int i = 0; i < 2; ++i)
      #pragma unroll
      for (int j = 0; j < 4; ++j) acc[i][j] = zero;
  }

  for (int kk = 0; kk < 256; kk += 64) {            // four K stages of 64
    if (kk) __syncthreads();                        // protect LDS before refill

    // ---- fill Ws: 128 rows(o) x 64 k; coalesced b128 reads, b64 LDS stores
    #pragma unroll
    for (int i = 0; i < 8; ++i) {
      int idx = tid + i * 256;                      // 0..2047 float4 units
      int row = idx >> 4;                           // 16 float4 per row
      int c4  = idx & 15;
      const float4 f = *reinterpret_cast<const float4*>(
          &w[(size_t)(oBase + row) * 256 + kk + c4 * 4]);
      *reinterpret_cast<us4*>(&WsLds[row][c4 * 4]) = cvt4_bf16(f.x, f.y, f.z, f.w);
    }

    // ---- fill Xs: transpose 4x4 f32 blocks in registers (x's k-dim strided in global)
    #pragma unroll
    for (int i = 0; i < 2; ++i) {
      int blk = tid + i * 256;                      // 0..511 blocks (16 kblk x 32 nblk)
      int kb4 = blk >> 5;
      int n4  = blk & 31;
      const float* px = &x[xBatch + (size_t)(kk + kb4 * 4) * 1024 + nBase + n4 * 4];
      float4 f0 = *reinterpret_cast<const float4*>(px);
      float4 f1 = *reinterpret_cast<const float4*>(px + 1024);
      float4 f2 = *reinterpret_cast<const float4*>(px + 2048);
      float4 f3 = *reinterpret_cast<const float4*>(px + 3072);
      *reinterpret_cast<us4*>(&XsLds[n4 * 4 + 0][kb4 * 4]) = cvt4_bf16(f0.x, f1.x, f2.x, f3.x);
      *reinterpret_cast<us4*>(&XsLds[n4 * 4 + 1][kb4 * 4]) = cvt4_bf16(f0.y, f1.y, f2.y, f3.y);
      *reinterpret_cast<us4*>(&XsLds[n4 * 4 + 2][kb4 * 4]) = cvt4_bf16(f0.z, f1.z, f2.z, f3.z);
      *reinterpret_cast<us4*>(&XsLds[n4 * 4 + 3][kb4 * 4]) = cvt4_bf16(f0.w, f1.w, f2.w, f3.w);
    }

    __syncthreads();

    // ---- prefetch next K-stage into L2/WGP$ while WMMAs run (global_prefetch_b8)
    if (kk + 64 < 256) {
      __builtin_prefetch(&w[(size_t)(oBase + (tid >> 1)) * 256 + kk + 64 + (tid & 1) * 32], 0, 1);
      __builtin_prefetch(&x[xBatch + (size_t)(kk + 64 + (tid >> 2)) * 1024 + nBase + (tid & 3) * 32], 0, 1);
    }

    // ---- compute: 2 k-steps of 32, 2x4 WMMA per wave per step
    #pragma unroll
    for (int kb = 0; kb < 64; kb += 32) {
      v16bf afrag[2], bfrag[4];
      #pragma unroll
      for (int ni = 0; ni < 2; ++ni) {
        const int rn = wn * 32 + ni * 16 + lrow;    // A row (n), per ISA A-layout
        Bits32B t;
        t.lo = *reinterpret_cast<const uint4*>(&XsLds[rn][kb + half * 8]);
        t.hi = *reinterpret_cast<const uint4*>(&XsLds[rn][kb + 16 + half * 8]);
        afrag[ni] = __builtin_bit_cast(v16bf, t);
      }
      #pragma unroll
      for (int oi = 0; oi < 4; ++oi) {
        const int ro = wo * 64 + oi * 16 + lrow;    // B column (o), per ISA B-layout
        Bits32B t;
        t.lo = *reinterpret_cast<const uint4*>(&WsLds[ro][kb + half * 8]);
        t.hi = *reinterpret_cast<const uint4*>(&WsLds[ro][kb + 16 + half * 8]);
        bfrag[oi] = __builtin_bit_cast(v16bf, t);
      }
      #pragma unroll
      for (int ni = 0; ni < 2; ++ni)
        #pragma unroll
        for (int oi = 0; oi < 4; ++oi)
          acc[ni][oi] = __builtin_amdgcn_wmma_f32_16x16x32_bf16(
              false, afrag[ni], false, bfrag[oi],
              (short)0, acc[ni][oi], false, false);
    }
  }

  // ---- epilogue: fused depth-to-space, staged through LDS for b128 global stores.
  // Each wave's 32n x 64o tile == 2048 consecutive output floats (2 image rows).
  __syncthreads();   // tiles dead; obuf aliases them

  const int h  = (nBase + wn * 32) >> 5;           // image block-row, fixed per wave
  const int r0 = (oBase + wo * 64) >> 5;           // first of 2 r values for this wave

  #pragma unroll
  for (int p = 0; p < 2; ++p) {                    // one 1024-float output row per pass
    #pragma unroll
    for (int ni = 0; ni < 2; ++ni) {
      #pragma unroll
      for (int qq = 0; qq < 2; ++qq) {
        const int oi   = p * 2 + qq;
        const int cRel = qq * 16 + lrow;           // 0..31
        #pragma unroll
        for (int v = 0; v < 8; ++v) {
          const int wRel = ni * 16 + half * 8 + v; // 0..31
          obuf[wave * 1024 + wRel * 32 + cRel] = acc[ni][oi][v];
        }
      }
    }
    __syncthreads();
    const size_t rowBase = ((size_t)b * 1024 + h * 32 + r0 + p) << 10;
    #pragma unroll
    for (int it = 0; it < 8; ++it) {               // 8 x (32 lanes x 16B) = 512B/instr
      float4 vv = *reinterpret_cast<const float4*>(&obuf[wave * 1024 + it * 128 + lane * 4]);
      *reinterpret_cast<float4*>(&out[rowBase + it * 128 + lane * 4]) = vv;
    }
    __syncthreads();
  }
}

extern "C" void kernel_launch(void* const* d_in, const int* in_sizes, int n_in,
                              void* d_out, int out_size, void* d_ws, size_t ws_size,
                              hipStream_t stream) {
  (void)in_sizes; (void)n_in; (void)d_ws; (void)ws_size; (void)out_size;
  const float* x = (const float*)d_in[0];   // [32,256,32,32] f32
  const float* w = (const float*)d_in[1];   // [1024,256] f32
  float* out = (float*)d_out;               // [32,1,1024,1024] f32

  dim3 grid(8, 8, 32);    // (o-tiles of 128, n-tiles of 128 per batch, batch)
  dim3 block(256);        // 8 waves
  conv1x1_d2s_wmma_bf16<<<grid, block, 0, stream>>>(x, w, out);
}
